// SemanticTikTokVQ_26439818674907
// MI455X (gfx1250) — matrix operand order repaired and checked by reading, not compiled, not run on hardware
//
#include <hip/hip_runtime.h>
#include <hip/hip_bf16.h>

#define N_TOK 32768
#define H_DIM 2048
#define D_DIM 256
#define K_CB  4096

typedef __attribute__((ext_vector_type(16))) __bf16 v16bf;
typedef __attribute__((ext_vector_type(8)))  __bf16 v8bf;
typedef __attribute__((ext_vector_type(8)))  float  v8f;

static __device__ inline v16bf make_frag(const __bf16* p0, const __bf16* p1) {
    v8bf a = *(const v8bf*)p0;
    v8bf b = *(const v8bf*)p1;
    v16bf r;
#pragma unroll
    for (int i = 0; i < 8; ++i) { r[i] = a[i]; r[8 + i] = b[i]; }
    return r;
}

static __device__ inline v8f wmma_bf16(v16bf a, v16bf b, v8f c) {
    return __builtin_amdgcn_wmma_f32_16x16x32_bf16(
        /*neg_a=*/false, a, /*neg_b=*/false, b,
        /*c_mod=*/(short)0, c, /*reuse_a=*/false, /*reuse_b=*/false);
}

// CDNA5 async global->LDS copy (ASYNCcnt-tracked), 16 bytes per lane.
static __device__ inline void async_b128(unsigned lds_byte, const void* gaddr) {
    asm volatile("global_load_async_to_lds_b128 %0, %1, off"
                 :: "v"(lds_byte), "v"(gaddr) : "memory");
}
static __device__ inline void wait_async0() {
    asm volatile("s_wait_asynccnt 0x0" ::: "memory");
}

// ---------------------------------------------------------------- utilities
__global__ void cvt_kernel(const float* __restrict__ in, __bf16* __restrict__ out, int n) {
    int i = blockIdx.x * blockDim.x + threadIdx.x;
    if (i < n) out[i] = (__bf16)in[i];
}

__global__ __launch_bounds__(256) void cnorm_kernel(const float* __restrict__ cb,
                                                    float* __restrict__ cnorm) {
    int lane = threadIdx.x & 31;
    int row  = (blockIdx.x * blockDim.x + threadIdx.x) >> 5;
    if (row >= K_CB) return;
    const float* cr = cb + (size_t)row * D_DIM;
    float s = 0.f;
    for (int d = lane; d < D_DIM; d += 32) s += cr[d] * cr[d];
#pragma unroll
    for (int off = 16; off; off >>= 1) s += __shfl_xor(s, off);
    if (lane == 0) cnorm[row] = s;
}

__global__ __launch_bounds__(256) void lse_kernel(const float* __restrict__ prior,
                                                  float* __restrict__ lse) {
    __shared__ float red[256];
    int tid = threadIdx.x;
    float m = -3.4e38f;
    for (int i = tid; i < K_CB; i += 256) m = fmaxf(m, prior[i]);
    red[tid] = m; __syncthreads();
    for (int s = 128; s; s >>= 1) { if (tid < s) red[tid] = fmaxf(red[tid], red[tid + s]); __syncthreads(); }
    float mx = red[0]; __syncthreads();
    float sum = 0.f;
    for (int i = tid; i < K_CB; i += 256) sum += expf(prior[i] - mx);
    red[tid] = sum; __syncthreads();
    for (int s = 128; s; s >>= 1) { if (tid < s) red[tid] += red[tid + s]; __syncthreads(); }
    if (tid == 0) lse[0] = mx + logf(red[0]);
}

__global__ void zero_kernel(float* accum) { if (threadIdx.x < 2) accum[threadIdx.x] = 0.f; }

// ------------------------------------------------- GEMM1: z = embed @ pre_w^T
// grid = N/16 blocks, 256 threads (8 waves). Block computes a 16x256 z tile.
// The 16x128 embed chunk is staged fp32->bf16 into a double-buffered LDS tile
// by the staging pass (cvt once per element); fragments are pure ds_load_b128.
// One barrier per chunk (8 WMMAs/wave); next chunk's global loads overlap WMMA.
#define K_CHUNK 128
#define N_CHUNK (H_DIM / K_CHUNK)
__global__ __launch_bounds__(256) void gemm1_kernel(const float* __restrict__ embed,
                                                    const __bf16* __restrict__ prew,
                                                    const float* __restrict__ pre_b,
                                                    float* __restrict__ zf,
                                                    __bf16* __restrict__ zb) {
    __shared__ __bf16 ab[2][16 * K_CHUNK];    // 2 x 4 KB
    const int tid  = threadIdx.x;
    const int lane = tid & 31;
    const int wid  = tid >> 5;
    const int row0 = blockIdx.x * 16;
    const int d0   = wid * 32;

    const int l16  = lane & 15;
    const int half = lane >> 4;               // 0 or 1
    const int c    = half << 4;               // K sub-offset for A frags

    // staging geometry: 8 contiguous fp32 per thread per chunk
    const int srow = tid >> 4;
    const int scol = (tid & 15) * 8;
    const float* gsrc = embed + (size_t)(row0 + srow) * H_DIM + scol;
    __bf16* sdst0 = &ab[0][srow * K_CHUNK + scol];
    __bf16* sdst1 = &ab[1][srow * K_CHUNK + scol];

    const __bf16* b0base = prew + (size_t)(d0 + l16) * H_DIM + (half << 4);
    const __bf16* b1base = prew + (size_t)(d0 + 16 + l16) * H_DIM + (half << 4);

    v8f acc0 = {}; v8f acc1 = {};
    v8f fstage = *(const v8f*)gsrc;          // chunk 0 in flight
    for (int t = 0; t < N_CHUNK; ++t) {
        // convert + store chunk t
        {
            v8bf hb;
#pragma unroll
            for (int i = 0; i < 8; ++i) hb[i] = (__bf16)fstage[i];
            *(v8bf*)((t & 1) ? sdst1 : sdst0) = hb;
        }
        __syncthreads();
        if (t + 1 < N_CHUNK)                  // overlap next chunk's loads with WMMA
            fstage = *(const v8f*)(gsrc + (t + 1) * K_CHUNK);
        const __bf16* arow = &ab[t & 1][l16 * K_CHUNK + c];
        const int k0 = t * K_CHUNK;
        __builtin_prefetch(b0base + k0 + K_CHUNK, 0, 1);
#pragma unroll
        for (int s = 0; s < K_CHUNK / 32; ++s) {
            v16bf af = make_frag(arow + s * 32, arow + s * 32 + 8);
            v16bf b0 = make_frag(b0base + k0 + s * 32, b0base + k0 + s * 32 + 8);
            v16bf b1 = make_frag(b1base + k0 + s * 32, b1base + k0 + s * 32 + 8);
            acc0 = wmma_bf16(af, b0, acc0);
            acc1 = wmma_bf16(af, b1, acc1);
        }
    }
    const int mBase = half << 3;
#pragma unroll
    for (int j = 0; j < 8; ++j) {
        int m = mBase + j;
        size_t o = (size_t)(row0 + m) * D_DIM + d0 + l16;
        float v0 = acc0[j] + pre_b[d0 + l16];
        float v1 = acc1[j] + pre_b[d0 + 16 + l16];
        zf[o] = v0;         zf[o + 16] = v1;
        zb[o] = (__bf16)v0; zb[o + 16] = (__bf16)v1;
    }
}

// ------------------------------------- VQ: argmin_k ||c_k||^2 - 2 z.c_k via WMMA
// grid = N/128, 256 threads; each wave owns 16 rows, scans all K=4096 codes.
// Codebook tiles (16 codes x 256 dims bf16 = 8 KB) are shared by all 8 waves:
// staged via async-to-LDS, double-buffered, copy of tile t+1 overlaps WMMA on t.
__global__ __launch_bounds__(256) void vq_kernel(const __bf16* __restrict__ zb,
                                                 const __bf16* __restrict__ cb,
                                                 const float* __restrict__ cnorm,
                                                 int* __restrict__ idx32,
                                                 long long* __restrict__ idx64) {
    __shared__ float  cn[K_CB];               // 16 KB
    __shared__ __bf16 bt[2][16 * D_DIM];      // 2 x 8 KB
    const int tid  = threadIdx.x;
    const int lane = tid & 31;
    const int wid  = tid >> 5;
    for (int i = tid; i < K_CB; i += 256) cn[i] = cnorm[i];

    const int row0 = blockIdx.x * 128 + wid * 16;
    const int l16  = lane & 15;
    const int half = lane >> 4;
    const int c    = half << 4;

    v16bf a[8];
    const __bf16* zrow = zb + (size_t)(row0 + l16) * D_DIM;
#pragma unroll
    for (int kk = 0; kk < 8; ++kk)
        a[kk] = make_frag(zrow + kk * 32 + c, zrow + kk * 32 + c + 8);

    const char* cbby = (const char*)cb;
    const unsigned lbase = (unsigned)(size_t)(&bt[0][0]) + tid * 32;
    // prologue: stage tile 0 into buffer 0 (2 x b128 per thread = 8 KB)
    async_b128(lbase,      cbby + tid * 32);
    async_b128(lbase + 16, cbby + tid * 32 + 16);

    float best[8]; int bidx[8];
#pragma unroll
    for (int j = 0; j < 8; ++j) { best[j] = 3.4e38f; bidx[j] = 0; }

    const int NT = K_CB / 16;
    for (int t = 0; t < NT; ++t) {
        wait_async0();                        // tile t resident (per-wave)
        __syncthreads();                      // ... for every wave
        if (t + 1 < NT) {                     // uniform branch: kick tile t+1
            unsigned ld = lbase + (unsigned)(((t + 1) & 1) * (16 * D_DIM * 2));
            const char* gs = cbby + (size_t)(t + 1) * (16 * D_DIM * 2) + tid * 32;
            async_b128(ld,      gs);
            async_b128(ld + 16, gs + 16);
        }
        const __bf16* cbp = &bt[t & 1][l16 * D_DIM + (half << 4)];
        v8f acc = {};
#pragma unroll
        for (int kk = 0; kk < 8; ++kk) {
            v16bf b = make_frag(cbp + kk * 32, cbp + kk * 32 + 8);
            acc = wmma_bf16(a[kk], b, acc);
        }
        int kcol = t * 16 + l16;
        float cv = cn[kcol];
#pragma unroll
        for (int j = 0; j < 8; ++j) {
            float dist = cv - 2.0f * acc[j];
            if (dist < best[j]) { best[j] = dist; bidx[j] = kcol; }
        }
    }
    // reduce across the 16 column-lanes of each half (wave32 shuffles)
#pragma unroll
    for (int off = 1; off < 16; off <<= 1) {
#pragma unroll
        for (int j = 0; j < 8; ++j) {
            float ov = __shfl_xor(best[j], off);
            int   oi = __shfl_xor(bidx[j], off);
            if (ov < best[j] || (ov == best[j] && oi < bidx[j])) { best[j] = ov; bidx[j] = oi; }
        }
    }
    if (l16 == 0) {
        int mBase = half << 3;
#pragma unroll
        for (int j = 0; j < 8; ++j) {
            int row = row0 + mBase + j;
            idx32[row] = bidx[j];
            idx64[row] = (long long)bidx[j];
        }
    }
}

// ----------------------- losses: sum (z_q - z)^2 and rate numerator, atomics
__global__ __launch_bounds__(256) void loss_kernel(const float* __restrict__ zf,
                                                   const float* __restrict__ cbf,
                                                   const int* __restrict__ idx32,
                                                   const float* __restrict__ prior,
                                                   const float* __restrict__ lse,
                                                   float* __restrict__ accum) {
    int tid  = blockIdx.x * blockDim.x + threadIdx.x;
    int lane = threadIdx.x & 31;
    int row  = tid >> 5;
    if (row >= N_TOK) return;
    int id = idx32[row];
    const float* z  = zf  + (size_t)row * D_DIM;
    const float* cq = cbf + (size_t)id * D_DIM;
    float s = 0.f;
    for (int d = lane; d < D_DIM; d += 32) { float df = cq[d] - z[d]; s += df * df; }
#pragma unroll
    for (int off = 16; off; off >>= 1) s += __shfl_xor(s, off);
    if (lane == 0) {
        atomicAdd(&accum[0], s);
        atomicAdd(&accum[1], -(prior[id] - lse[0]));
    }
}

// --------------------------------- GEMM2: embed_hat = gather(cb,idx) @ post_w^T
// grid = N/128, 256 threads; wave owns 16 rows, loops full H=2048 output cols.
// post_w tiles (16 cols x 256 dims bf16 = 8 KB) shared by all waves: async
// double-buffered staging identical to vq_kernel.
__global__ __launch_bounds__(256) void gemm2_kernel(const __bf16* __restrict__ cb,
                                                    const int* __restrict__ idx32,
                                                    const __bf16* __restrict__ postw,
                                                    const float* __restrict__ post_b,
                                                    float* __restrict__ out) {
    __shared__ __bf16 wt[2][16 * D_DIM];      // 2 x 8 KB
    const int tid  = threadIdx.x;
    const int lane = tid & 31;
    const int wid  = tid >> 5;
    const int row0 = blockIdx.x * 128 + wid * 16;
    const int l16  = lane & 15;
    const int half = lane >> 4;
    const int c    = half << 4;

    const int id = idx32[row0 + l16];
    const __bf16* q = cb + (size_t)id * D_DIM;
    v16bf a[8];
#pragma unroll
    for (int kk = 0; kk < 8; ++kk)
        a[kk] = make_frag(q + kk * 32 + c, q + kk * 32 + c + 8);

    const char* pwby = (const char*)postw;
    const unsigned lbase = (unsigned)(size_t)(&wt[0][0]) + tid * 32;
    async_b128(lbase,      pwby + tid * 32);
    async_b128(lbase + 16, pwby + tid * 32 + 16);

    const int kOff = half << 4;
    const int NT = H_DIM / 16;
    for (int t = 0; t < NT; ++t) {
        wait_async0();
        __syncthreads();
        if (t + 1 < NT) {
            unsigned ld = lbase + (unsigned)(((t + 1) & 1) * (16 * D_DIM * 2));
            const char* gs = pwby + (size_t)(t + 1) * (16 * D_DIM * 2) + tid * 32;
            async_b128(ld,      gs);
            async_b128(ld + 16, gs + 16);
        }
        const __bf16* bp = &wt[t & 1][l16 * D_DIM + kOff];
        v8f acc = {};
#pragma unroll
        for (int kk = 0; kk < 8; ++kk) {
            v16bf b = make_frag(bp + kk * 32, bp + kk * 32 + 8);
            acc = wmma_bf16(a[kk], b, acc);
        }
        const int h0 = t * 16;
        float bb = post_b[h0 + l16];
        int mBase = half << 3;
#pragma unroll
        for (int j = 0; j < 8; ++j) {
            int m = mBase + j;
            out[(size_t)(row0 + m) * H_DIM + h0 + l16] = acc[j] + bb;
        }
    }
}

__global__ void scalars_kernel(const float* __restrict__ accum, float* __restrict__ tail) {
    if (threadIdx.x == 0) {
        tail[0] = accum[1] / 0.6931471805599453f;                         // rate_bits
        tail[1] = 1.25f * accum[0] / (float)((size_t)N_TOK * D_DIM);      // vq_loss
    }
}

// ---------------------------------------------------------------------------
extern "C" void kernel_launch(void* const* d_in, const int* in_sizes, int n_in,
                              void* d_out, int out_size, void* d_ws, size_t ws_size,
                              hipStream_t stream) {
    const float* embed  = (const float*)d_in[0];
    const float* pre_w  = (const float*)d_in[1];
    const float* pre_b  = (const float*)d_in[2];
    const float* cbook  = (const float*)d_in[3];
    const float* post_w = (const float*)d_in[4];
    const float* post_b = (const float*)d_in[5];
    const float* prior  = (const float*)d_in[6];

    char* w = (char*)d_ws;
    size_t off = 0;
    float*  zf     = (float*)(w + off);  off += (size_t)N_TOK * D_DIM * 4;
    __bf16* zb     = (__bf16*)(w + off); off += (size_t)N_TOK * D_DIM * 2;
    __bf16* prewb  = (__bf16*)(w + off); off += (size_t)D_DIM * H_DIM * 2;
    __bf16* cbb    = (__bf16*)(w + off); off += (size_t)K_CB * D_DIM * 2;
    __bf16* postwb = (__bf16*)(w + off); off += (size_t)H_DIM * D_DIM * 2;
    float*  cnorm  = (float*)(w + off);  off += (size_t)K_CB * 4;
    int*    idx32  = (int*)(w + off);    off += (size_t)N_TOK * 4;
    float*  lse    = (float*)(w + off);  off += 16;
    float*  accum  = (float*)(w + off);  off += 16;

    float*     embed_hat = (float*)d_out;
    long long* idx64     = (long long*)((char*)d_out + (size_t)N_TOK * H_DIM * 4);
    float*     tail      = (float*)((char*)d_out + (size_t)N_TOK * H_DIM * 4 + (size_t)N_TOK * 8);

    // weight conversions
    cvt_kernel<<<(D_DIM * H_DIM + 255) / 256, 256, 0, stream>>>(pre_w, prewb, D_DIM * H_DIM);
    cvt_kernel<<<(K_CB * D_DIM + 255) / 256, 256, 0, stream>>>(cbook, cbb, K_CB * D_DIM);
    cvt_kernel<<<(H_DIM * D_DIM + 255) / 256, 256, 0, stream>>>(post_w, postwb, H_DIM * D_DIM);
    cnorm_kernel<<<(K_CB * 32) / 256, 256, 0, stream>>>(cbook, cnorm);
    lse_kernel<<<1, 256, 0, stream>>>(prior, lse);
    zero_kernel<<<1, 32, 0, stream>>>(accum);

    // pre projection (WMMA bf16, pipelined LDS staging)
    gemm1_kernel<<<N_TOK / 16, 256, 0, stream>>>(embed, prewb, pre_b, zf, zb);
    // nearest-code search (WMMA bf16 + argmin, async double-buffered codebook)
    vq_kernel<<<N_TOK / 128, 256, 0, stream>>>(zb, cbb, cnorm, idx32, idx64);
    // losses
    loss_kernel<<<(N_TOK * 32) / 256, 256, 0, stream>>>(zf, cbook, idx32, prior, lse, accum);
    // post projection from gathered codes (WMMA bf16, async double-buffered B)
    gemm2_kernel<<<N_TOK / 128, 256, 0, stream>>>(cbb, idx32, postwb, post_b, embed_hat);
    scalars_kernel<<<1, 32, 0, stream>>>(accum, tail);
}
